// PointNetSetAbstractionMsg_2972117368915
// MI455X (gfx1250) — matrix operand order repaired and checked by reading, not compiled, not run on hardware
//
#include <hip/hip_runtime.h>
#include <hip/hip_bf16.h>

typedef _Float16 v16h __attribute__((ext_vector_type(16)));
typedef _Float16 v8h  __attribute__((ext_vector_type(8)));
typedef float    v8f  __attribute__((ext_vector_type(8)));

#define BB 16
#define NN 4096
#define SS 512

// ---------------------------------------------------------------------------
// FPS: one block per batch. dist[] lives in LDS; 512 dependent argmax steps.
// Tie-break = lowest index to match jnp.argmax (first max).
// ---------------------------------------------------------------------------
__global__ __launch_bounds__(256) void fps_kernel(const float* __restrict__ xyz,
        int* __restrict__ fps_idx, float* __restrict__ nxyz,
        float* __restrict__ out_xyz)
{
    const int b = blockIdx.x, tid = threadIdx.x;
    __shared__ float dist[NN];
    __shared__ float rv[256];
    __shared__ int   ri[256];
    const float* xb = xyz + (size_t)b*3*NN;
    for (int i = tid; i < NN; i += 256) dist[i] = 1e10f;
    __syncthreads();
    int far = 0;
    for (int it = 0; it < SS; ++it) {
        float cx = xb[far], cy = xb[NN+far], cz = xb[2*NN+far];
        if (tid == 0) {
            fps_idx[b*SS+it] = far;
            nxyz[((size_t)b*SS+it)*3+0] = cx;
            nxyz[((size_t)b*SS+it)*3+1] = cy;
            nxyz[((size_t)b*SS+it)*3+2] = cz;
            out_xyz[(size_t)b*3*SS + 0*SS + it] = cx;
            out_xyz[(size_t)b*3*SS + 1*SS + it] = cy;
            out_xyz[(size_t)b*3*SS + 2*SS + it] = cz;
        }
        float best = -1.0f; int bi = 0;
        for (int i = tid; i < NN; i += 256) {
            float dx = xb[i]-cx, dy = xb[NN+i]-cy, dz = xb[2*NN+i]-cz;
            float d  = dx*dx + dy*dy + dz*dz;
            float nd = fminf(dist[i], d);
            dist[i] = nd;
            if (nd > best) { best = nd; bi = i; }
        }
        rv[tid] = best; ri[tid] = bi;
        __syncthreads();
        for (int s2 = 128; s2 > 0; s2 >>= 1) {
            if (tid < s2) {
                if (rv[tid+s2] > rv[tid] ||
                    (rv[tid+s2] == rv[tid] && ri[tid+s2] < ri[tid])) {
                    rv[tid] = rv[tid+s2]; ri[tid] = ri[tid+s2];
                }
            }
            __syncthreads();
        }
        far = ri[0];
        __syncthreads();
    }
}

// ---------------------------------------------------------------------------
// Ball query: one wave32 per (b,s). ballot+popc compaction => first K in
// index order; pad tail with first in-radius index.
// ---------------------------------------------------------------------------
__global__ __launch_bounds__(256) void ball_kernel(const float* __restrict__ xyz,
        const float* __restrict__ nxyz, int* __restrict__ gidx,
        int K, float r2)
{
    const int wid  = blockIdx.x*8 + (threadIdx.x >> 5);
    const int lane = threadIdx.x & 31;
    const int b = wid / SS;
    const float* xb = xyz + (size_t)b*3*NN;
    const float cx = nxyz[(size_t)wid*3+0];
    const float cy = nxyz[(size_t)wid*3+1];
    const float cz = nxyz[(size_t)wid*3+2];
    int* g = gidx + (size_t)wid*K;
    int count = 0, firstn = 0;
    bool haveFirst = false;
    for (int base = 0; base < NN && count < K; base += 32) {
        int n = base + lane;
        float dx = xb[n]-cx, dy = xb[NN+n]-cy, dz = xb[2*NN+n]-cz;
        float d = dx*dx + dy*dy + dz*dz;
        bool inR = d <= r2;
        unsigned m = __builtin_amdgcn_ballot_w32(inR);
        int pos = count + __popc(m & ((1u << lane) - 1u));
        if (inR && pos < K) g[pos] = n;
        if (!haveFirst && m) { firstn = base + __ffs(m) - 1; haveFirst = true; }
        count += __popc(m);
    }
    int total = count < K ? count : K;
    for (int p = total + lane; p < K; p += 32) g[p] = firstn;
}

// ---------------------------------------------------------------------------
// Grouping: h0[b,s,k, 0:3]=points[idx], 3:6=xyz[idx]-new_xyz, 6:32=0 (f16),
// written as 4 x b128 stores.
// ---------------------------------------------------------------------------
__global__ __launch_bounds__(256) void group_kernel(const float* __restrict__ points,
        const float* __restrict__ xyz, const float* __restrict__ nxyz,
        const int* __restrict__ gidx, _Float16* __restrict__ h0, int K, int Mtot)
{
    int t = blockIdx.x*256 + threadIdx.x;
    if (t >= Mtot) return;
    int bs = t / K;
    int b  = bs / SS;
    int idx = gidx[t];
    const float* pb = points + (size_t)b*3*NN;
    const float* xb = xyz    + (size_t)b*3*NN;
    v8h f0;
    f0[0] = (_Float16)pb[idx];
    f0[1] = (_Float16)pb[NN+idx];
    f0[2] = (_Float16)pb[2*NN+idx];
    f0[3] = (_Float16)(xb[idx]      - nxyz[(size_t)bs*3+0]);
    f0[4] = (_Float16)(xb[NN+idx]   - nxyz[(size_t)bs*3+1]);
    f0[5] = (_Float16)(xb[2*NN+idx] - nxyz[(size_t)bs*3+2]);
    f0[6] = (_Float16)0.0f;
    f0[7] = (_Float16)0.0f;
    v8h z = {};
    _Float16* d = h0 + (size_t)t*32;
    *(v8h*)(d)      = f0;
    *(v8h*)(d + 8)  = z;
    *(v8h*)(d + 16) = z;
    *(v8h*)(d + 24) = z;
}

// ---------------------------------------------------------------------------
// Weight conversion: W (Cout x Cin, f32) -> f16 B-tile swizzle, zero-padded
// to Kpad. Per (ktile,ntile,lane): 16 contiguous halves = lane's WMMA B chunk:
// b[i] = W^T[k0 + (lane>>4)*16 + i][ntile*16 + (lane&15)].
// ---------------------------------------------------------------------------
__global__ void wconv_kernel(const float* __restrict__ W, _Float16* __restrict__ Wb,
                             int Cin, int Cout, int Kpad)
{
    int t = blockIdx.x*256 + threadIdx.x;
    int NT = Cout >> 4, KT = Kpad >> 5;
    if (t >= KT*NT*32) return;
    int lane = t & 31;
    int nt = (t >> 5) % NT;
    int kt = (t >> 5) / NT;
    int col = nt*16 + (lane & 15);
    int kb  = kt*32 + (lane >> 4)*16;
    _Float16* dst = Wb + ((size_t)(kt*NT + nt)*32 + lane)*16;
    #pragma unroll
    for (int i = 0; i < 16; ++i) {
        int k = kb + i;
        dst[i] = (k < Cin) ? (_Float16)W[(size_t)col*Cin + k] : (_Float16)0.0f;
    }
}

// ---------------------------------------------------------------------------
// WMMA GEMM: D(MxCout,f16) = A(MxCin,f16) * Wt(CinxCout). 8 waves/block,
// one 16-row tile per wave; A held in VGPRs across all N-tiles; B from LDS.
// Epilogue: wave-local LDS transpose of the 16x16 fragment so each lane
// emits one contiguous global_store_b128 (instead of 8 scalar b16 stores).
// Same-wave DS ops are in-order -> no barrier needed.
// ---------------------------------------------------------------------------
template<int KT>
__global__ __launch_bounds__(256) void gemm_kernel(const _Float16* __restrict__ A,
        const _Float16* __restrict__ Wb, _Float16* __restrict__ D,
        int Cin, int Cout)
{
    const int NT = Cout >> 4;
    __shared__ alignas(32) _Float16 sWb[3*8*32*16];   // up to 96x128 f16 = 24KB
    __shared__ alignas(32) _Float16 sOut[8][16*16];   // per-wave 512B transpose tile
    const int tid = threadIdx.x;
    {
        const int n32 = KT * NT * 256;                // dwords of Wb used
        const unsigned* src = (const unsigned*)Wb;
        unsigned* dst = (unsigned*)sWb;
        for (int i = tid; i < n32; i += 256) dst[i] = src[i];
    }
    __syncthreads();
    const int wave = tid >> 5, lane = tid & 31;
    const int half = lane >> 4;
    const int m0   = (blockIdx.x*8 + wave) * 16;
    const int row  = m0 + (lane & 15);

    v16h a[KT];
    #pragma unroll
    for (int kt = 0; kt < KT; ++kt) {
        // A 16x32 f16 layout: lane holds K = [k0+half*16 .. +15] (matches B packing)
        const _Float16* pr = A + (size_t)row*Cin + kt*32 + half*16;
        v8h lo = *(const v8h*)pr;
        v8h hi = *(const v8h*)(pr + 8);
        a[kt] = __builtin_shufflevector(lo, hi, 0,1,2,3,4,5,6,7,8,9,10,11,12,13,14,15);
    }
    _Float16* tp = &sOut[wave][0];
    for (int nt = 0; nt < NT; ++nt) {
        v8f acc = {0.f,0.f,0.f,0.f,0.f,0.f,0.f,0.f};
        #pragma unroll
        for (int kt = 0; kt < KT; ++kt) {
            const v16h b = *(const v16h*)(sWb + ((size_t)(kt*NT + nt)*32 + lane)*16);
            acc = __builtin_amdgcn_wmma_f32_16x16x32_f16(false, a[kt], false, b,
                                                         (short)0, acc, false, false);
        }
        // fragment -> LDS (row-major 16x16 f16 tile)
        #pragma unroll
        for (int r = 0; r < 8; ++r) {
            tp[(half*8 + r)*16 + (lane & 15)] = (_Float16)acc[r];
        }
        // contiguous half-row readback -> single b128 global store
        const int rr = lane & 15;
        v8h rowv = *(const v8h*)(tp + rr*16 + half*8);
        *(v8h*)(D + (size_t)(m0 + rr)*Cout + nt*16 + half*8) = rowv;
    }
}

// ---------------------------------------------------------------------------
// BN statistics, bit-deterministic (no float atomics):
// stage 1: (C/8)*1024 threads; each owns an 8-channel group and a fixed
//          strided row set (v8h loads). psum layout stays psum[q*C + c].
// stage 2: one block per channel reduces the 1024 partials in fixed order.
// ---------------------------------------------------------------------------
__global__ __launch_bounds__(256) void stats_partial_kernel(const _Float16* __restrict__ h,
        float* __restrict__ psum, float* __restrict__ psq, int M, int C)
{
    int tid = blockIdx.x*256 + threadIdx.x;   // T = (C/8)*1024
    int cg = C >> 3;
    int g = tid % cg;
    int q = tid / cg;
    int c0 = g*8;
    float s[8], ss[8];
    #pragma unroll
    for (int j = 0; j < 8; ++j) { s[j] = 0.f; ss[j] = 0.f; }
    for (int row = q; row < M; row += 1024) {
        v8h v = *(const v8h*)(h + (size_t)row*C + c0);
        #pragma unroll
        for (int j = 0; j < 8; ++j) {
            float f = (float)v[j];
            s[j] += f; ss[j] += f*f;
        }
    }
    #pragma unroll
    for (int j = 0; j < 8; ++j) {
        psum[(size_t)q*C + c0 + j] = s[j];
        psq [(size_t)q*C + c0 + j] = ss[j];
    }
}

__global__ __launch_bounds__(256) void bnstats_kernel(const float* __restrict__ psum,
        const float* __restrict__ psq, const float* __restrict__ gamma,
        const float* __restrict__ beta, float* __restrict__ scale,
        float* __restrict__ shift, int M, int C)
{
    int c = blockIdx.x, t = threadIdx.x;
    __shared__ float ls[256], lq[256];
    float s = 0.f, q = 0.f;
    for (int i = t; i < 1024; i += 256) {
        s += psum[(size_t)i*C + c];
        q += psq[(size_t)i*C + c];
    }
    ls[t] = s; lq[t] = q;
    __syncthreads();
    for (int st = 128; st > 0; st >>= 1) {
        if (t < st) { ls[t] += ls[t+st]; lq[t] += lq[t+st]; }
        __syncthreads();
    }
    if (t == 0) {
        float inv = 1.0f / (float)M;
        float mu  = ls[0]*inv;
        float var = lq[0]*inv - mu*mu;
        float sc  = gamma[c] * rsqrtf(var + 1e-5f);
        scale[c] = sc;
        shift[c] = beta[c] - mu*sc;
    }
}

// 8 halves per thread; groups never cross the channel wrap (C % 8 == 0).
__global__ __launch_bounds__(256) void bnrelu_kernel(_Float16* __restrict__ h,
        const float* __restrict__ scale, const float* __restrict__ shift,
        size_t total8, int C)
{
    size_t i = (size_t)blockIdx.x*256 + threadIdx.x;
    if (i >= total8) return;
    size_t base = i*8;
    int c0 = (int)(base % C);
    v8h v = *(const v8h*)(h + base);
    v8h r;
    #pragma unroll
    for (int j = 0; j < 8; ++j) {
        float f = (float)v[j]*scale[c0+j] + shift[c0+j];
        r[j] = (_Float16)fmaxf(f, 0.f);
    }
    *(v8h*)(h + base) = r;
}

// ---------------------------------------------------------------------------
// Max over K samples -> new_points slice (B, 320, S). One thread per
// (b,s, 8-channel group); v8h loads down the K axis.
// ---------------------------------------------------------------------------
__global__ __launch_bounds__(256) void maxk_kernel(const _Float16* __restrict__ h,
        float* __restrict__ out_np, int K, int C, int coff)
{
    int t = blockIdx.x*256 + threadIdx.x;     // total = B*S*(C/8)
    int cg = C >> 3;
    int g = t % cg;
    int bs = t / cg;
    int b = bs / SS, s = bs % SS;
    int c0 = g*8;
    const _Float16* hp = h + ((size_t)bs*K)*C + c0;
    float m[8];
    #pragma unroll
    for (int j = 0; j < 8; ++j) m[j] = -1e30f;
    for (int k = 0; k < K; ++k) {
        v8h v = *(const v8h*)(hp + (size_t)k*C);
        #pragma unroll
        for (int j = 0; j < 8; ++j) m[j] = fmaxf(m[j], (float)v[j]);
    }
    float* op = out_np + (size_t)b*320*SS + (size_t)(coff+c0)*SS + s;
    #pragma unroll
    for (int j = 0; j < 8; ++j) op[(size_t)j*SS] = m[j];
}

// ---------------------------------------------------------------------------
// Host orchestration. Workspace layout (bytes), ~453 MB total:
//   0         fps_idx  int[8192]
//   32768     nxyz     float[8192*3]
//   131072    gidx     int[8192*128]
//   4325376   Wb       f16[96*128]
//   4349952   scale/shift float[256]
//   4352000   psum     float[131072]
//   4876288   psq      float[131072]
//   5400576   bufX     f16[1048576*96]   (192 MB)
//   206727168 bufY     f16[1048576*128]  (256 MB)
// ---------------------------------------------------------------------------
extern "C" void kernel_launch(void* const* d_in, const int* in_sizes, int n_in,
                              void* d_out, int out_size, void* d_ws, size_t ws_size,
                              hipStream_t stream)
{
    const float* xyz    = (const float*)d_in[0];
    const float* points = (const float*)d_in[1];
    char* ws = (char*)d_ws;
    int*      fps_idx = (int*)(ws + 0);
    float*    nxyz    = (float*)(ws + 32768);
    int*      gidx    = (int*)(ws + 131072);
    _Float16* Wb      = (_Float16*)(ws + 4325376);
    float*    scale   = (float*)(ws + 4349952);
    float*    shift   = scale + 128;
    float*    psum    = (float*)(ws + 4352000);
    float*    psq     = (float*)(ws + 4876288);
    _Float16* bufX    = (_Float16*)(ws + 5400576);
    _Float16* bufY    = (_Float16*)(ws + 206727168);

    float* out_xyz = (float*)d_out;
    float* out_np  = out_xyz + (size_t)BB*3*SS;

    fps_kernel<<<BB, 256, 0, stream>>>(xyz, fps_idx, nxyz, out_xyz);

    const int   KS[3]     = {16, 32, 128};
    const float R2[3]     = {0.01f, 0.04f, 0.16f};
    const int   MLP[3][3] = {{32,32,64},{64,64,128},{64,96,128}};
    const int   COFF[3]   = {0, 64, 192};

    for (int br = 0; br < 3; ++br) {
        const int K = KS[br];
        const int M = BB*SS*K;
        ball_kernel<<<BB*SS/8, 256, 0, stream>>>(xyz, nxyz, gidx, K, R2[br]);
        group_kernel<<<M/256, 256, 0, stream>>>(points, xyz, nxyz, gidx, bufX, K, M);

        _Float16* hin = bufX;
        _Float16* hout = bufY;
        int CinAct = 6, CinPad = 32;
        for (int j = 0; j < 3; ++j) {
            const int Cout = MLP[br][j];
            const int base = 2 + (br*3 + j)*4;     // params order: W, b, gamma, beta
            const float* W     = (const float*)d_in[base+0];
            const float* gamma = (const float*)d_in[base+2];
            const float* beta  = (const float*)d_in[base+3];
            const int KT = CinPad >> 5;
            const int NT = Cout >> 4;
            wconv_kernel<<<(KT*NT*32 + 255)/256, 256, 0, stream>>>(W, Wb, CinAct, Cout, CinPad);
            switch (KT) {
                case 1:  gemm_kernel<1><<<M/128, 256, 0, stream>>>(hin, Wb, hout, CinPad, Cout); break;
                case 2:  gemm_kernel<2><<<M/128, 256, 0, stream>>>(hin, Wb, hout, CinPad, Cout); break;
                default: gemm_kernel<3><<<M/128, 256, 0, stream>>>(hin, Wb, hout, CinPad, Cout); break;
            }
            stats_partial_kernel<<<(Cout/8)*4, 256, 0, stream>>>(hout, psum, psq, M, Cout);
            bnstats_kernel<<<Cout, 256, 0, stream>>>(psum, psq, gamma, beta, scale, shift, M, Cout);
            size_t total8 = (size_t)M*Cout/8;
            bnrelu_kernel<<<(unsigned)(total8/256), 256, 0, stream>>>(hout, scale, shift, total8, Cout);
            _Float16* tmp = hin; hin = hout; hout = tmp;
            CinAct = Cout; CinPad = Cout;
        }
        const int Clast = MLP[br][2];
        maxk_kernel<<<BB*SS*(Clast/8)/256, 256, 0, stream>>>(hin, out_np, K, Clast, COFF[br]);
    }
}
